// UNetDecoderBlock_25598005084691
// MI455X (gfx1250) — matrix operand order repaired and checked by reading, not compiled, not run on hardware
//
#include <hip/hip_runtime.h>

// ---------------- types ----------------
typedef __attribute__((ext_vector_type(16))) __bf16 v16bf;
typedef __attribute__((ext_vector_type(8)))  float  v8f;

union FragAB { uint4 q[2]; v16bf v; };

static __device__ __forceinline__ unsigned short f2bf(float f) {
    union { float f; unsigned u; } v; v.f = f;
    unsigned r = v.u + 0x7FFFu + ((v.u >> 16) & 1u);   // round-to-nearest-even
    return (unsigned short)(r >> 16);
}

static __device__ __forceinline__ v8f wmma_bf16(v16bf a, v16bf b, v8f c) {
    return __builtin_amdgcn_wmma_f32_16x16x32_bf16(false, a, false, b, (short)0, c, false, false);
}

#define S_OUT 128
#define TSIZE (S_OUT * S_OUT * S_OUT)
#define CIN 128
#define COUT 64
#define WBYTES (27 * 4096 * 2)        // permuted weights, bytes (221184)
#define WQUADS (WBYTES / 16)          // as uint4 (13824)

// ---------------- kernels ----------------

// reset hash tables to -1; zero BN stats (6 * 64 floats)
__global__ void k_init(int* __restrict__ table_s, int* __restrict__ table_o,
                       float* __restrict__ stats) {
    int i = blockIdx.x * blockDim.x + threadIdx.x;
    if (i < TSIZE) { table_s[i] = -1; table_o[i] = -1; }
    if (i < 6 * 64) stats[i] = 0.0f;
}

__global__ void k_scatter_skip(const int* __restrict__ skip_cords,
                               int* __restrict__ table_s, int NS) {
    int i = blockIdx.x * blockDim.x + threadIdx.x;
    if (i >= NS) return;
    int x = skip_cords[i * 3 + 0], y = skip_cords[i * 3 + 1], z = skip_cords[i * 3 + 2];
    table_s[(x * S_OUT + y) * S_OUT + z] = i;
}

// permute one stage's weights [27][64][64] f32 -> bf16 in WMMA B-fragment order:
// wp[k][n][c][lane][e] = W[k][ c*32 + (lane>>4)*8 + (e&7) + (e>>3)*16 ][ n*16 + (lane&15) ]
__global__ void k_permw(const float* __restrict__ w, unsigned short* __restrict__ wp) {
    int t = blockIdx.x * blockDim.x + threadIdx.x;
    if (t >= 27 * 4096) return;
    int e    = t & 15;
    int lane = (t >> 4) & 31;
    int c    = (t >> 9) & 1;
    int n    = (t >> 10) & 3;
    int k    = t >> 12;
    int K    = c * 32 + (lane >> 4) * 8 + (e & 7) + (e >> 3) * 16;
    int col  = n * 16 + (lane & 15);
    wp[t] = f2bf(w[k * 4096 + K * 64 + col]);
}

// upsample (fp32) + skip add + table_o scatter; one block (64 threads) per output row m
__global__ void k_upsample(const float* __restrict__ x_feats,
                           const float* __restrict__ w_up,
                           const float* __restrict__ skip_feats,
                           const int* __restrict__ cords,
                           const int* __restrict__ table_s,
                           int* __restrict__ table_o,
                           unsigned short* __restrict__ feats0, int M) {
    __shared__ float xrow[CIN];
    int m = blockIdx.x;
    if (m >= M) return;
    int c = threadIdx.x;               // output channel 0..63
    int n = m >> 3, o = m & 7;
    xrow[c]      = x_feats[(size_t)n * CIN + c];
    xrow[c + 64] = x_feats[(size_t)n * CIN + c + 64];
    __syncthreads();
    const float* wrow = w_up + (size_t)(7 - o) * (CIN * COUT);
    float acc = 0.0f;
    #pragma unroll 8
    for (int j = 0; j < CIN; ++j) acc += xrow[j] * wrow[j * COUT + c];
    int cx = 2 * cords[n * 3 + 0] + ((o >> 2) & 1);
    int cy = 2 * cords[n * 3 + 1] + ((o >> 1) & 1);
    int cz = 2 * cords[n * 3 + 2] + (o & 1);
    int lin = (cx * S_OUT + cy) * S_OUT + cz;
    int j = table_s[lin];
    if (j >= 0) acc += skip_feats[(size_t)j * COUT + c];
    feats0[(size_t)m * COUT + c] = f2bf(acc);
    if (c == 0) table_o[lin] = m;
}

// neighbor table nbr[k][m], k in meshgrid(-1..1, 'ij') order
__global__ void k_neighbors(const int* __restrict__ cords,
                            const int* __restrict__ table_o,
                            int* __restrict__ nbr, int M) {
    int m = blockIdx.x * blockDim.x + threadIdx.x;
    if (m >= M) return;
    int n = m >> 3, o = m & 7;
    int cx = 2 * cords[n * 3 + 0] + ((o >> 2) & 1);
    int cy = 2 * cords[n * 3 + 1] + ((o >> 1) & 1);
    int cz = 2 * cords[n * 3 + 2] + (o & 1);
    int k = 0;
    for (int dx = -1; dx <= 1; ++dx)
      for (int dy = -1; dy <= 1; ++dy)
        for (int dz = -1; dz <= 1; ++dz) {
            int x = cx + dx, y = cy + dy, z = cz + dz;
            int r = -1;
            if (((unsigned)x < S_OUT) & ((unsigned)y < S_OUT) & ((unsigned)z < S_OUT))
                r = table_o[(x * S_OUT + y) * S_OUT + z];
            nbr[(size_t)k * M + m] = r;
            ++k;
        }
}

// gather-GEMM submanifold conv.
// 256 threads = 8 waves; weights staged in LDS (216 KB of gfx1250's 320 KB WGP LDS).
// Each wave owns 32 rows (two 16-row A tiles sharing B fragments) x all 64 columns.
__global__ void k_conv_wmma(const unsigned short* __restrict__ fin,
                            const int* __restrict__ nbr,
                            const unsigned short* __restrict__ wperm,
                            float* __restrict__ yout,
                            float* __restrict__ ssum, float* __restrict__ ssq, int M) {
    extern __shared__ char smem[];
    uint4* wlds = (uint4*)smem;                       // WQUADS uint4
    float* lsum = (float*)(smem + WBYTES);
    float* lsq  = lsum + COUT;

    int tid = threadIdx.x;
    // stage permuted weights global -> LDS (one-time per block)
    const uint4* wq = (const uint4*)wperm;
    for (int i = tid; i < WQUADS; i += 256) wlds[i] = wq[i];
    if (tid < COUT) { lsum[tid] = 0.0f; lsq[tid] = 0.0f; }
    __syncthreads();

    int lane = tid & 31, wave = tid >> 5;
    int m0 = blockIdx.x * 256 + wave * 32;            // two 16-row tiles per wave
    int g = lane >> 4;
    int row0 = m0 + (lane & 15);
    int row1 = row0 + 16;

    v8f acc[2][4] = {};
    for (int k = 0; k < 27; ++k) {
        const size_t kb = (size_t)k * M;
        int idx0 = nbr[kb + row0];
        int idx1 = nbr[kb + row1];
        FragAB a00, a01, a10, a11;                    // [tile][K-chunk]
        if (idx0 >= 0) {
            const uint4* p = (const uint4*)(fin + (size_t)idx0 * COUT);
            a00.q[0] = p[g];     a00.q[1] = p[2 + g];
            a01.q[0] = p[4 + g]; a01.q[1] = p[6 + g];
        } else {
            uint4 z; z.x = z.y = z.z = z.w = 0u;
            a00.q[0] = a00.q[1] = a01.q[0] = a01.q[1] = z;
        }
        if (idx1 >= 0) {
            const uint4* p = (const uint4*)(fin + (size_t)idx1 * COUT);
            a10.q[0] = p[g];     a10.q[1] = p[2 + g];
            a11.q[0] = p[4 + g]; a11.q[1] = p[6 + g];
        } else {
            uint4 z; z.x = z.y = z.z = z.w = 0u;
            a10.q[0] = a10.q[1] = a11.q[0] = a11.q[1] = z;
        }
        const uint4* wb = wlds + (size_t)k * 512;     // 512 uint4 per k
        #pragma unroll
        for (int n = 0; n < 4; ++n) {
            FragAB b0, b1;                            // [K-chunk], shared by both tiles
            int base = n * 128 + lane * 2;
            b0.q[0] = wb[base];      b0.q[1] = wb[base + 1];
            b1.q[0] = wb[base + 64]; b1.q[1] = wb[base + 65];
            acc[0][n] = wmma_bf16(a00.v, b0.v, acc[0][n]);
            acc[0][n] = wmma_bf16(a01.v, b1.v, acc[0][n]);
            acc[1][n] = wmma_bf16(a10.v, b0.v, acc[1][n]);
            acc[1][n] = wmma_bf16(a11.v, b1.v, acc[1][n]);
        }
    }

    int nl = lane & 15;
    #pragma unroll
    for (int t = 0; t < 2; ++t) {
        int mrow = m0 + t * 16 + g * 8;
        #pragma unroll
        for (int n = 0; n < 4; ++n) {
            int col = n * 16 + nl;
            float s = 0.0f, q = 0.0f;
            #pragma unroll
            for (int r = 0; r < 8; ++r) {
                float v = acc[t][n][r];
                yout[(size_t)(mrow + r) * COUT + col] = v;
                s += v; q += v * v;
            }
            atomicAdd(&lsum[col], s);
            atomicAdd(&lsq[col], q);
        }
    }
    __syncthreads();
    if (tid < COUT) { atomicAdd(&ssum[tid], lsum[tid]); atomicAdd(&ssq[tid], lsq[tid]); }
}

// BN + activation. mode 0: lrelu(.05) -> x1f (f32) + xb (bf16)
//                  mode 1: lrelu(.05) -> xb (bf16)
//                  mode 2: bn + residual, lrelu(.333) -> outf (f32)
__global__ void k_bn_apply(const float* __restrict__ y,
                           const float* __restrict__ ssum, const float* __restrict__ ssq,
                           const float* __restrict__ gamma, const float* __restrict__ beta,
                           float* __restrict__ x1f, unsigned short* __restrict__ xb,
                           float* __restrict__ outf, const float* __restrict__ resf,
                           float invM, int total, int mode) {
    int i = blockIdx.x * blockDim.x + threadIdx.x;
    if (i >= total) return;
    int c = i & (COUT - 1);
    float mean = ssum[c] * invM;
    float var  = ssq[c] * invM - mean * mean;
    float sc   = rsqrtf(var + 1e-4f) * gamma[c];
    float z = (y[i] - mean) * sc + beta[c];
    if (mode == 2) {
        z += resf[i];
        outf[i] = (z >= 0.0f) ? z : 0.333f * z;
    } else {
        z = (z >= 0.0f) ? z : 0.05f * z;
        xb[i] = f2bf(z);
        if (mode == 0) x1f[i] = z;
    }
}

// ---------------- launcher ----------------
extern "C" void kernel_launch(void* const* d_in, const int* in_sizes, int n_in,
                              void* d_out, int out_size, void* d_ws, size_t ws_size,
                              hipStream_t stream) {
    const float* x_feats    = (const float*)d_in[0];
    const float* skip_feats = (const float*)d_in[1];
    const float* w_up       = (const float*)d_in[2];
    const float* w1         = (const float*)d_in[3];
    const float* w2         = (const float*)d_in[4];
    const float* w3         = (const float*)d_in[5];
    const float* g1 = (const float*)d_in[6],  *b1 = (const float*)d_in[7];
    const float* g2 = (const float*)d_in[8],  *b2 = (const float*)d_in[9];
    const float* g3 = (const float*)d_in[10], *b3 = (const float*)d_in[11];
    const int* cords      = (const int*)d_in[12];
    const int* skip_cords = (const int*)d_in[13];

    const int NX = in_sizes[12] / 3;
    const int M  = NX * 8;
    const int NS = in_sizes[13] / 3;

    // workspace carve-up (256B aligned)
    size_t off = 0;
    auto alloc = [&](size_t sz) -> void* {
        void* p = (void*)((char*)d_ws + off);
        off += (sz + 255) & ~(size_t)255;
        return p;
    };
    int*            table_s = (int*)alloc((size_t)TSIZE * 4);
    int*            table_o = (int*)alloc((size_t)TSIZE * 4);
    int*            nbr     = (int*)alloc((size_t)27 * M * 4);
    unsigned short* feats0  = (unsigned short*)alloc((size_t)M * COUT * 2);
    unsigned short* xb      = (unsigned short*)alloc((size_t)M * COUT * 2);
    float*          yA      = (float*)alloc((size_t)M * COUT * 4);
    float*          x1f     = (float*)alloc((size_t)M * COUT * 4);
    unsigned short* wp1     = (unsigned short*)alloc((size_t)WBYTES);
    unsigned short* wp2     = (unsigned short*)alloc((size_t)WBYTES);
    unsigned short* wp3     = (unsigned short*)alloc((size_t)WBYTES);
    float*          stats   = (float*)alloc((size_t)6 * COUT * 4);
    float *sum1 = stats,       *sq1 = stats + 64;
    float *sum2 = stats + 128, *sq2 = stats + 192;
    float *sum3 = stats + 256, *sq3 = stats + 320;

    const int total = M * COUT;
    const float invM = 1.0f / (float)M;
    const int convGrid = M / 256;                       // 32 rows/wave * 8 waves
    const size_t convLds = (size_t)WBYTES + 2 * COUT * sizeof(float);

    k_init<<<(TSIZE + 255) / 256, 256, 0, stream>>>(table_s, table_o, stats);
    k_scatter_skip<<<(NS + 255) / 256, 256, 0, stream>>>(skip_cords, table_s, NS);
    k_permw<<<(27 * 4096 + 255) / 256, 256, 0, stream>>>(w1, wp1);
    k_permw<<<(27 * 4096 + 255) / 256, 256, 0, stream>>>(w2, wp2);
    k_permw<<<(27 * 4096 + 255) / 256, 256, 0, stream>>>(w3, wp3);
    k_upsample<<<M, 64, 0, stream>>>(x_feats, w_up, skip_feats, cords,
                                     table_s, table_o, feats0, M);
    k_neighbors<<<(M + 255) / 256, 256, 0, stream>>>(cords, table_o, nbr, M);

    // conv1 -> bn/lrelu -> x1 (f32 + bf16)
    k_conv_wmma<<<convGrid, 256, convLds, stream>>>(feats0, nbr, wp1, yA, sum1, sq1, M);
    k_bn_apply<<<(total + 255) / 256, 256, 0, stream>>>(yA, sum1, sq1, g1, b1,
        x1f, xb, nullptr, nullptr, invM, total, 0);
    // conv2 -> bn/lrelu -> x2 (bf16)
    k_conv_wmma<<<convGrid, 256, convLds, stream>>>(xb, nbr, wp2, yA, sum2, sq2, M);
    k_bn_apply<<<(total + 255) / 256, 256, 0, stream>>>(yA, sum2, sq2, g2, b2,
        nullptr, xb, nullptr, nullptr, invM, total, 1);
    // conv3 -> bn + residual -> out
    k_conv_wmma<<<convGrid, 256, convLds, stream>>>(xb, nbr, wp3, yA, sum3, sq3, M);
    k_bn_apply<<<(total + 255) / 256, 256, 0, stream>>>(yA, sum3, sq3, g3, b3,
        nullptr, nullptr, (float*)d_out, x1f, invM, total, 2);
}
